// Ernie4_5_VLMoeBlock_44289702756737
// MI455X (gfx1250) — matrix operand order repaired and checked by reading, not compiled, not run on hardware
//
#include <hip/hip_runtime.h>
#include <hip/hip_bf16.h>

// ---------------------------------------------------------------------------
// Ernie4.5-VL MoE block for gfx1250 (MI455X), bf16 WMMA pipeline.
// LDS-staged, double-buffered B tiles via GLOBAL_LOAD_ASYNC_TO_LDS_B128.
// ---------------------------------------------------------------------------

#define T_TOK   1024
#define DIM     1024
#define NEXP    8
#define F_TEXT  512
#define F_VIS   256
#define F_SH    1024

typedef __attribute__((ext_vector_type(16))) __bf16        v16bf;
typedef __attribute__((ext_vector_type(8)))  float         v8f;
typedef __attribute__((ext_vector_type(4)))  unsigned int  v4u;
typedef int v4i_vs __attribute__((vector_size(16)));   // matches builtin param type

union FragBF {
    v16bf v;
    v4u   d[2];   // two 16-byte chunks
};

// ------------------ optional gfx1250 async-to-LDS support -------------------
#if defined(__has_builtin)
#  if __has_builtin(__builtin_amdgcn_global_load_async_to_lds_b128)
#    define HAVE_ASYNC_LDS 1
#  endif
#  if __has_builtin(__builtin_amdgcn_s_wait_asynccnt)
#    define HAVE_WAIT_ASYNC 1
#  endif
#endif

#if defined(HAVE_ASYNC_LDS) && defined(HAVE_WAIT_ASYNC)
#  define USE_ASYNC_LDS 1
#endif

// Copy 16 bytes global -> LDS (per lane). Async path uses the gfx1250
// GLOBAL_LOAD_ASYNC_TO_LDS_B128 instruction (tracked by ASYNCcnt).
__device__ __forceinline__ void stage16(const unsigned short* g, unsigned short* l) {
#if defined(USE_ASYNC_LDS)
    // generic LDS pointer: low 32 bits are the LDS byte offset (ISA aperture rule)
    __builtin_amdgcn_global_load_async_to_lds_b128(
        (__attribute__((address_space(1))) v4i_vs*)(size_t)g,
        (__attribute__((address_space(3))) v4i_vs*)(unsigned int)(size_t)l,
        0, 0);
#else
    *(v4u*)l = *(const v4u*)g;
#endif
}

__device__ __forceinline__ void stage_wait() {
#if defined(USE_ASYNC_LDS)
    __builtin_amdgcn_s_wait_asynccnt(0);
#endif
}

__device__ __forceinline__ unsigned short f32_to_bf16(float f) {
    union { float f; unsigned int u; } v; v.f = f;
    unsigned int r = v.u + 0x7FFFu + ((v.u >> 16) & 1u);   // round-to-nearest-even
    return (unsigned short)(r >> 16);
}

// --------------------------- fp32 -> bf16 copy ------------------------------
__global__ void __launch_bounds__(256)
convert_bf16_kernel(const float* __restrict__ src, unsigned short* __restrict__ dst, int n) {
    int i = blockIdx.x * 256 + threadIdx.x;
    if (i < n) dst[i] = f32_to_bf16(src[i]);
}

// ---------------- fp32 [B,R,C] -> bf16 [B,C,R] (transpose) ------------------
__global__ void __launch_bounds__(256)
transpose_bf16_kernel(const float* __restrict__ src, unsigned short* __restrict__ dst,
                      int R, int C) {
    __shared__ float tile[32][33];
    const size_t plane = (size_t)R * C;
    const float*     s = src + (size_t)blockIdx.z * plane;
    unsigned short*  d = dst + (size_t)blockIdx.z * plane;
    const int c0 = blockIdx.x * 32, r0 = blockIdx.y * 32;
    const int tx = threadIdx.x & 31, ty = threadIdx.x >> 5;   // 32 x 8
    #pragma unroll
    for (int i = 0; i < 32; i += 8) {
        int r = r0 + ty + i, c = c0 + tx;
        tile[ty + i][tx] = (r < R && c < C) ? s[(size_t)r * C + c] : 0.0f;
    }
    __syncthreads();
    #pragma unroll
    for (int i = 0; i < 32; i += 8) {
        int c = c0 + ty + i, r = r0 + tx;
        if (c < C && r < R) d[(size_t)c * R + r] = f32_to_bf16(tile[tx][ty + i]);
    }
}

// ------------------------------- router -------------------------------------
__global__ void __launch_bounds__(256)
router_kernel(const float* __restrict__ x, const int* __restrict__ ttype,
              const float* __restrict__ trw, const float* __restrict__ tbias,
              const float* __restrict__ vrw, const float* __restrict__ vbias,
              float* __restrict__ comb_t, float* __restrict__ comb_v,
              float* __restrict__ logits_out) {
    int t = blockIdx.x * 256 + threadIdx.x;
    if (t >= T_TOK) return;
    const float* xr = x + (size_t)t * DIM;

    float lt[NEXP], lv[NEXP];
    #pragma unroll
    for (int e = 0; e < NEXP; ++e) { lt[e] = 0.f; lv[e] = 0.f; }
    for (int k = 0; k < DIM; ++k) {
        float xv = xr[k];
        #pragma unroll
        for (int e = 0; e < NEXP; ++e) {
            lt[e] += xv * trw[e * DIM + k];
            lv[e] += xv * vrw[e * DIM + k];
        }
    }
    const bool vis = (ttype[t] != 0);

    #pragma unroll
    for (int e = 0; e < NEXP; ++e)
        logits_out[(size_t)t * NEXP + e] = vis ? lv[e] : lt[e];

    auto route = [&](const float* l, const float* bias, float* comb, bool active) {
        float mx = l[0];
        #pragma unroll
        for (int e = 1; e < NEXP; ++e) mx = fmaxf(mx, l[e]);
        float p[NEXP], s = 0.f;
        #pragma unroll
        for (int e = 0; e < NEXP; ++e) { p[e] = __expf(l[e] - mx); s += p[e]; }
        float inv = 1.0f / s;
        #pragma unroll
        for (int e = 0; e < NEXP; ++e) p[e] *= inv;
        int i0 = 0; float b0 = -1e30f;
        #pragma unroll
        for (int e = 0; e < NEXP; ++e) { float v = p[e] + bias[e]; if (v > b0) { b0 = v; i0 = e; } }
        int i1 = 0; float b1 = -1e30f;
        #pragma unroll
        for (int e = 0; e < NEXP; ++e) {
            if (e == i0) continue;
            float v = p[e] + bias[e]; if (v > b1) { b1 = v; i1 = e; }
        }
        float w0 = p[i0], w1 = p[i1];
        float sw = fmaxf(w0 + w1, 1e-12f);
        w0 /= sw; w1 /= sw;
        #pragma unroll
        for (int e = 0; e < NEXP; ++e) comb[(size_t)t * NEXP + e] = 0.f;
        if (active) {
            comb[(size_t)t * NEXP + i0] = w0;
            comb[(size_t)t * NEXP + i1] = w1;
        }
    };
    route(lt, tbias, comb_t, !vis);
    route(lv, vbias, comb_v,  vis);
}

// ----------------- fragment loaders (CDNA5 WMMA bf16 layouts) ---------------
// A 16x32 bf16: lanes 0-15 row M=lane, K 0-7 & 16-23; lanes 16-31 same rows,
//               K 8-15 & 24-31  -> two contiguous 16B chunks per lane.
// B 32x16 bf16 (from B^T [N,K] tile staged in LDS): column N = lane&15,
//               lanes 0-15 K 0-7 & 16-23, lanes 16-31 K 8-15 & 24-31.
__device__ __forceinline__ FragBF load_a_frag(const unsigned short* aRow, int k0) {
    FragBF f;
    f.d[0] = *(const v4u*)(aRow + k0);
    f.d[1] = *(const v4u*)(aRow + k0 + 16);
    return f;
}
__device__ __forceinline__ FragBF load_lds_frag(const unsigned short* l) {
    FragBF f;
    const v4u* p = (const v4u*)l;
    f.d[0] = p[0];     // ds_load_b128
    f.d[1] = p[2];     // ds_load_b128
    return f;
}

__device__ __forceinline__ v8f wmma_bf16(const FragBF& a, const FragBF& b, v8f c) {
    return __builtin_amdgcn_wmma_f32_16x16x32_bf16(
        false, a.v, false, b.v, (short)0, c, false, false);
}

// ------------------------ gate+up SwiGLU GEMM (WMMA) ------------------------
// A: x_bf [M,K] bf16; BgT/BuT: [N,K] bf16 (pre-transposed); H: [M,N] bf16.
// Block = 8 waves = 128M x 64N; B tiles (64N x 32K) staged in LDS, 2x buffered.
__global__ void __launch_bounds__(256)
gateup_wmma_kernel(const unsigned short* __restrict__ A,
                   const unsigned short* __restrict__ BgT,
                   const unsigned short* __restrict__ BuT,
                   unsigned short* __restrict__ H,
                   int M, int N, int K) {
    __shared__ unsigned short lbg[2][64 * 32];
    __shared__ unsigned short lbu[2][64 * 32];

    const int tid  = threadIdx.x;
    const int lane = tid & 31;
    const int wave = tid >> 5;
    const int m0 = blockIdx.x * 128 + wave * 16;
    const int n0 = blockIdx.y * 64;
    const int rlo = lane & 15;
    const int hi  = lane >> 4;

    // cooperative staging map: 256 threads x 16B = one 64x32 bf16 tile
    const int srow   = tid >> 2;          // 0..63
    const int schunk = (tid & 3) * 8;     // element offset 0,8,16,24
    const unsigned short* gBg = BgT + (size_t)(n0 + srow) * K + schunk;
    const unsigned short* gBu = BuT + (size_t)(n0 + srow) * K + schunk;
    const int soff = srow * 32 + schunk;

    v8f accg[4] = {}, accu[4] = {};
    const unsigned short* aRow = A + (size_t)(m0 + rlo) * K + hi * 8;

    // one K-step of work: batch all 8 fragment loads, then 8 WMMAs back-to-back
    auto compute = [&](int b, int kc) {
        FragBF a = load_a_frag(aRow, kc);
        FragBF bg[4], bu[4];
        #pragma unroll
        for (int nt = 0; nt < 4; ++nt) {
            const int foff = (nt * 16 + rlo) * 32 + hi * 8;
            bg[nt] = load_lds_frag(&lbg[b][foff]);
            bu[nt] = load_lds_frag(&lbu[b][foff]);
        }
        #pragma unroll
        for (int nt = 0; nt < 4; ++nt) {
            accg[nt] = wmma_bf16(a, bg[nt], accg[nt]);
            accu[nt] = wmma_bf16(a, bu[nt], accu[nt]);
        }
    };

    // prologue: stage first K tile into buffer 0
    stage16(gBg, &lbg[0][soff]);
    stage16(gBu, &lbu[0][soff]);

    int buf = 0, k0 = 0;
    for (; k0 + 32 < K; k0 += 32) {        // steady state: branch-free body
        stage_wait();                       // s_wait_asynccnt 0
        __syncthreads();                    // tile(buf) ready; tile(buf^1) free
        stage16(gBg + k0 + 32, &lbg[buf ^ 1][soff]);
        stage16(gBu + k0 + 32, &lbu[buf ^ 1][soff]);
        __builtin_prefetch((const void*)(aRow + k0 + 64), 0, 3);  // global_prefetch_b8
        compute(buf, k0);
        buf ^= 1;
    }
    stage_wait();                           // epilogue K-step (peeled)
    __syncthreads();
    compute(buf, k0);

    // epilogue: h = silu(g) * u, store bf16. C layout: (vgpr r, lane) ->
    // row m0 + r + 8*(lane>>4), col n0 + nt*16 + (lane&15)
    #pragma unroll
    for (int nt = 0; nt < 4; ++nt) {
        #pragma unroll
        for (int r = 0; r < 8; ++r) {
            float g = accg[nt][r], u = accu[nt][r];
            float sig = 1.0f / (1.0f + __expf(-g));
            float hval = g * sig * u;
            int m = m0 + r + 8 * hi;
            int n = n0 + nt * 16 + rlo;
            H[(size_t)m * N + n] = f32_to_bf16(hval);
        }
    }
}

// ---------------------- down-proj GEMM (WMMA) + combine ---------------------
// A: h [M,K] bf16; BT: wd^T [N,K] bf16; Out: [M,N] fp32.
// comb==nullptr -> weight 1.0; accumulate: 0 = write, 1 = add.
__global__ void __launch_bounds__(256)
down_wmma_kernel(const unsigned short* __restrict__ A,
                 const unsigned short* __restrict__ BT,
                 float* __restrict__ Out,
                 const float* __restrict__ comb, int expert, int accumulate,
                 int M, int N, int K) {
    __shared__ unsigned short lb[2][64 * 32];

    const int tid  = threadIdx.x;
    const int lane = tid & 31;
    const int wave = tid >> 5;
    const int m0 = blockIdx.x * 128 + wave * 16;
    const int n0 = blockIdx.y * 64;
    const int rlo = lane & 15;
    const int hi  = lane >> 4;

    const int srow   = tid >> 2;
    const int schunk = (tid & 3) * 8;
    const unsigned short* gB = BT + (size_t)(n0 + srow) * K + schunk;
    const int soff = srow * 32 + schunk;

    v8f acc[4] = {};
    const unsigned short* aRow = A + (size_t)(m0 + rlo) * K + hi * 8;

    auto compute = [&](int b, int kc) {
        FragBF a = load_a_frag(aRow, kc);
        FragBF bb[4];
        #pragma unroll
        for (int nt = 0; nt < 4; ++nt)
            bb[nt] = load_lds_frag(&lb[b][(nt * 16 + rlo) * 32 + hi * 8]);
        #pragma unroll
        for (int nt = 0; nt < 4; ++nt)
            acc[nt] = wmma_bf16(a, bb[nt], acc[nt]);
    };

    stage16(gB, &lb[0][soff]);

    int buf = 0, k0 = 0;
    for (; k0 + 32 < K; k0 += 32) {
        stage_wait();
        __syncthreads();
        stage16(gB + k0 + 32, &lb[buf ^ 1][soff]);
        __builtin_prefetch((const void*)(aRow + k0 + 64), 0, 3);
        compute(buf, k0);
        buf ^= 1;
    }
    stage_wait();
    __syncthreads();
    compute(buf, k0);

    float cr[8];
    #pragma unroll
    for (int r = 0; r < 8; ++r) {
        int m = m0 + r + 8 * hi;
        cr[r] = comb ? comb[(size_t)m * NEXP + expert] : 1.0f;
    }
    #pragma unroll
    for (int nt = 0; nt < 4; ++nt) {
        #pragma unroll
        for (int r = 0; r < 8; ++r) {
            int m = m0 + r + 8 * hi;
            int n = n0 + nt * 16 + rlo;
            size_t idx = (size_t)m * N + n;
            float v = cr[r] * acc[nt][r];
            if (accumulate) Out[idx] += v;
            else            Out[idx]  = v;
        }
    }
}

// ---------------------------------------------------------------------------
extern "C" void kernel_launch(void* const* d_in, const int* in_sizes, int n_in,
                              void* d_out, int out_size, void* d_ws, size_t ws_size,
                              hipStream_t stream) {
    (void)in_sizes; (void)n_in; (void)out_size;

    const float* x        = (const float*)d_in[0];
    const int*   ttype    = (const int*)  d_in[1];
    const float* trw      = (const float*)d_in[2];
    const float* tbias    = (const float*)d_in[3];
    const float* t_wg     = (const float*)d_in[4];
    const float* t_wu     = (const float*)d_in[5];
    const float* t_wd     = (const float*)d_in[6];
    const float* vrw      = (const float*)d_in[7];
    const float* vbias    = (const float*)d_in[8];
    const float* v_wg     = (const float*)d_in[9];
    const float* v_wu     = (const float*)d_in[10];
    const float* v_wd     = (const float*)d_in[11];
    const float* sh_wg    = (const float*)d_in[12];
    const float* sh_wu    = (const float*)d_in[13];
    const float* sh_wd    = (const float*)d_in[14];

    float* out    = (float*)d_out;                       // [T, D]
    float* logits = out + (size_t)T_TOK * DIM;           // [T, E]

    // ------------- workspace carve-up (256B aligned slices) ----------------
    char*  wsb = (char*)d_ws;
    size_t off = 0;
    auto alloc = [&](size_t bytes) -> char* {
        char* p = wsb + off;
        off = (off + bytes + 255) & ~(size_t)255;
        return p;
    };
    float*          comb_t = (float*)alloc((size_t)T_TOK * NEXP * 4);
    float*          comb_v = (float*)alloc((size_t)T_TOK * NEXP * 4);
    unsigned short* x_bf   = (unsigned short*)alloc((size_t)T_TOK * DIM * 2);
    unsigned short* h_bf   = (unsigned short*)alloc((size_t)T_TOK * F_SH * 2);
    unsigned short* shWgT  = (unsigned short*)alloc((size_t)F_SH * DIM * 2);
    unsigned short* shWuT  = (unsigned short*)alloc((size_t)F_SH * DIM * 2);
    unsigned short* shWdT  = (unsigned short*)alloc((size_t)DIM * F_SH * 2);
    unsigned short* tWgT   = (unsigned short*)alloc((size_t)NEXP * F_TEXT * DIM * 2);
    unsigned short* tWuT   = (unsigned short*)alloc((size_t)NEXP * F_TEXT * DIM * 2);
    unsigned short* tWdT   = (unsigned short*)alloc((size_t)NEXP * DIM * F_TEXT * 2);
    unsigned short* vWgT   = (unsigned short*)alloc((size_t)NEXP * F_VIS * DIM * 2);
    unsigned short* vWuT   = (unsigned short*)alloc((size_t)NEXP * F_VIS * DIM * 2);
    unsigned short* vWdT   = (unsigned short*)alloc((size_t)NEXP * DIM * F_VIS * 2);
    if (off > ws_size) return;   // workspace too small: bail deterministically

    // ----------------------- conversions / transposes ----------------------
    convert_bf16_kernel<<<(T_TOK * DIM + 255) / 256, 256, 0, stream>>>(x, x_bf, T_TOK * DIM);

    transpose_bf16_kernel<<<dim3(F_SH / 32, DIM / 32, 1),  256, 0, stream>>>(sh_wg, shWgT, DIM,  F_SH);
    transpose_bf16_kernel<<<dim3(F_SH / 32, DIM / 32, 1),  256, 0, stream>>>(sh_wu, shWuT, DIM,  F_SH);
    transpose_bf16_kernel<<<dim3(DIM / 32,  F_SH / 32, 1), 256, 0, stream>>>(sh_wd, shWdT, F_SH, DIM);
    transpose_bf16_kernel<<<dim3(F_TEXT / 32, DIM / 32, NEXP),    256, 0, stream>>>(t_wg, tWgT, DIM,    F_TEXT);
    transpose_bf16_kernel<<<dim3(F_TEXT / 32, DIM / 32, NEXP),    256, 0, stream>>>(t_wu, tWuT, DIM,    F_TEXT);
    transpose_bf16_kernel<<<dim3(DIM / 32,    F_TEXT / 32, NEXP), 256, 0, stream>>>(t_wd, tWdT, F_TEXT, DIM);
    transpose_bf16_kernel<<<dim3(F_VIS / 32, DIM / 32, NEXP),    256, 0, stream>>>(v_wg, vWgT, DIM,   F_VIS);
    transpose_bf16_kernel<<<dim3(F_VIS / 32, DIM / 32, NEXP),    256, 0, stream>>>(v_wu, vWuT, DIM,   F_VIS);
    transpose_bf16_kernel<<<dim3(DIM / 32,   F_VIS / 32, NEXP),  256, 0, stream>>>(v_wd, vWdT, F_VIS, DIM);

    // ------------------------------- router --------------------------------
    router_kernel<<<T_TOK / 256, 256, 0, stream>>>(x, ttype, trw, tbias, vrw, vbias,
                                                   comb_t, comb_v, logits);

    // ------------------------- shared experts (every token) ----------------
    gateup_wmma_kernel<<<dim3(T_TOK / 128, F_SH / 64), 256, 0, stream>>>(
        x_bf, shWgT, shWuT, h_bf, T_TOK, F_SH, DIM);
    down_wmma_kernel<<<dim3(T_TOK / 128, DIM / 64), 256, 0, stream>>>(
        h_bf, shWdT, out, nullptr, 0, /*accumulate=*/0, T_TOK, DIM, F_SH);

    // --------------------------- text sparse MoE ---------------------------
    for (int e = 0; e < NEXP; ++e) {
        const unsigned short* wg = tWgT + (size_t)e * F_TEXT * DIM;
        const unsigned short* wu = tWuT + (size_t)e * F_TEXT * DIM;
        const unsigned short* wd = tWdT + (size_t)e * DIM * F_TEXT;
        gateup_wmma_kernel<<<dim3(T_TOK / 128, F_TEXT / 64), 256, 0, stream>>>(
            x_bf, wg, wu, h_bf, T_TOK, F_TEXT, DIM);
        down_wmma_kernel<<<dim3(T_TOK / 128, DIM / 64), 256, 0, stream>>>(
            h_bf, wd, out, comb_t, e, /*accumulate=*/1, T_TOK, DIM, F_TEXT);
    }

    // -------------------------- vision sparse MoE --------------------------
    for (int e = 0; e < NEXP; ++e) {
        const unsigned short* wg = vWgT + (size_t)e * F_VIS * DIM;
        const unsigned short* wu = vWuT + (size_t)e * F_VIS * DIM;
        const unsigned short* wd = vWdT + (size_t)e * DIM * F_VIS;
        gateup_wmma_kernel<<<dim3(T_TOK / 128, F_VIS / 64), 256, 0, stream>>>(
            x_bf, wg, wu, h_bf, T_TOK, F_VIS, DIM);
        down_wmma_kernel<<<dim3(T_TOK / 128, DIM / 64), 256, 0, stream>>>(
            h_bf, wd, out, comb_v, e, /*accumulate=*/1, T_TOK, DIM, F_VIS);
    }
}
